// Attention_6743098655482
// MI455X (gfx1250) — matrix operand order repaired and checked by reading, not compiled, not run on hardware
//
#include <hip/hip_runtime.h>
#include <cstdint>
#include <cstddef>

// ---- CDNA5 WMMA vector types (wave32) ----
typedef _Float16 v16h __attribute__((ext_vector_type(16)));
typedef float    v8f  __attribute__((ext_vector_type(8)));

#define B_      64
#define H_      256
#define L_      8192
#define TWO_H   512
#define NTILE   64      // l-columns per block (4 waves x 16)
#define KP      128     // packed f16-pair dwords along K (=256 halves)
#define XSTRIDE 132     // KP + 4 pad dwords: 16B-aligned frags, de-conflicted banks

// tanh(x) = 1 - 2/(exp(2x)+1); saturates correctly at +-1 for large |x|
__device__ __forceinline__ float fast_tanhf(float x) {
    float e = __expf(2.0f * x);                       // v_mul + v_exp_f32
    return 1.0f - 2.0f * __builtin_amdgcn_rcpf(e + 1.0f);
}

struct __align__(16) BFrag { uint32_t u[8]; };        // 32B loaded as 2x ds_load_b128

// -------------------------------------------------------------------------
// Pack W[:, :H] (f32) into per-lane A-fragment layout for f16 16x16x32 WMMA.
// Layout in ws: [m_tile(16)][k_step(8)][lane(32)][8 dwords = 16 f16].
// Lane<16: row M = 16*mt+lane,   K offsets {0..15}
// Lane>=16: row M = 16*mt+lane-16, K offsets {16..31}
// -------------------------------------------------------------------------
__global__ void pack_w1_kernel(const float* __restrict__ W, uint32_t* __restrict__ w1p) {
    int tid  = blockIdx.x * blockDim.x + threadIdx.x;   // 0..4095
    int lane = tid & 31;
    int ks   = (tid >> 5) & 7;
    int mt   = tid >> 8;
    int m    = mt * 16 + (lane & 15);
    int hi   = lane >> 4;
    const float* wrow = W + (size_t)m * TWO_H;          // W1 row (first H columns)
    uint32_t* out = w1p + (size_t)tid * 8;
#pragma unroll
    for (int d = 0; d < 8; ++d) {
        int koff = hi * 16 + 2 * d;
        int k = ks * 32 + koff;
        union { _Float16 h[2]; uint32_t u; } p;
        p.h[0] = (_Float16)wrow[k];       // K even -> low 16 bits
        p.h[1] = (_Float16)wrow[k + 1];   // K odd  -> high 16 bits
        out[d] = p.u;
    }
}

// bias[b,h] = sum_k W[h, H+k] * mc[b,k]   (f32, tiny)
__global__ void bias_kernel(const float* __restrict__ W, const float* __restrict__ mc,
                            float* __restrict__ bias) {
    int tid = blockIdx.x * blockDim.x + threadIdx.x;    // 0..16383
    int b = tid >> 8;
    int h = tid & 255;
    const float* wrow = W + (size_t)h * TWO_H + H_;
    const float* mcr  = mc + (size_t)b * H_;
    float s = 0.f;
    for (int k = 0; k < H_; ++k) s = fmaf(wrow[k], mcr[k], s);
    bias[tid] = s;
}

// -------------------------------------------------------------------------
// Main fused kernel: scores[b,l] = sum_h v[h] * tanh( (W1 @ X_b)[h,l] + bias[b,h] )
// One block = one batch b and 64 l-columns; 4 waves, each owning 16 columns.
// -------------------------------------------------------------------------
__global__ __launch_bounds__(128) void attn_scores_kernel(
    const float* __restrict__ X,      // sn_hidden [B,H,L]
    const float* __restrict__ v,      // [H]
    const float* __restrict__ bias,   // [B,H]
    const v16h*  __restrict__ w1p,    // packed A fragments
    float*       __restrict__ scores) // d_out [B,L] (pre-softmax)
{
    __shared__ uint32_t xlds[NTILE * XSTRIDE];   // X tile, f16 pairs, [n][kpair]
    __shared__ float vs[H_];
    __shared__ float bs[H_];

    const int tile = blockIdx.x;        // 0..127
    const int b    = blockIdx.y;        // 0..63
    const int l0   = tile * NTILE;
    const int t    = threadIdx.x;       // 0..127

    vs[t]       = v[t];
    vs[t + 128] = v[t + 128];
    bs[t]       = bias[(size_t)b * H_ + t];
    bs[t + 128] = bias[(size_t)b * H_ + t + 128];

    // Cooperative load: f32 [k][l] coalesced along l -> f16-pair, transposed [n][k]
    {
        const int n   = t & (NTILE - 1);   // 0..63
        const int kpb = t >> 6;            // 0..1
        const float* xb = X + (size_t)b * H_ * L_ + l0 + n;
#pragma unroll 4
        for (int p = 0; p < KP / 2; ++p) {
            int kp = (p << 1) | kpb;
            int k  = kp << 1;
            float g0 = xb[(size_t)k * L_];
            float g1 = xb[(size_t)(k + 1) * L_];
            union { _Float16 h[2]; uint32_t u; } pk;
            pk.h[0] = (_Float16)g0;
            pk.h[1] = (_Float16)g1;
            xlds[n * XSTRIDE + kp] = pk.u;
        }
    }
    __syncthreads();

    const int lane    = t & 31;
    const int wave    = t >> 5;            // 0..3
    const int ncol    = lane & 15;
    const int hi      = lane >> 4;         // B-frag: lanes 16-31 hold K=16..31
    const int n_local = wave * 16 + ncol;

    const uint32_t* bbase = &xlds[n_local * XSTRIDE + hi * 8];
    float acc = 0.f;

#pragma unroll 1
    for (int mt = 0; mt < 16; ++mt) {
        v8f c = {};
        const v16h* afrag = w1p + (size_t)(mt * 8) * 32 + lane;
        if (mt < 15)  // pull next m-tile's A fragments toward the WGP
            __builtin_prefetch((const void*)(w1p + (size_t)((mt + 1) * 8) * 32 + lane), 0, 1);
#pragma unroll
        for (int ks = 0; ks < 8; ++ks) {
            v16h a = afrag[(size_t)ks * 32];
            union { BFrag s; v16h h; } bu;
            bu.s = *(const BFrag*)(bbase + ks * 16);
            c = __builtin_amdgcn_wmma_f32_16x16x32_f16(
                    false, a, false, bu.h, (short)0, c, false, false);
        }
        // C layout: VGPR i holds M = 16*mt + 8*hi + i, N = lane&15
#pragma unroll
        for (int i = 0; i < 8; ++i) {
            int m = mt * 16 + hi * 8 + i;
            acc = fmaf(vs[m], fast_tanhf(c[i] + bs[m]), acc);
        }
    }

    acc += __shfl_xor(acc, 16, 32);        // combine the two M-halves per column
    if (hi == 0)
        scores[(size_t)b * L_ + l0 + n_local] = acc;
}

// In-place softmax over each row of 8192 (recomputed every call -> replay-safe)
__global__ __launch_bounds__(256) void softmax_kernel(float* __restrict__ s) {
    const int b = blockIdx.x;
    const int t = threadIdx.x;
    float* row = s + (size_t)b * L_;
    float x[32];
    float mx = -3.402823466e38f;
#pragma unroll
    for (int j = 0; j < 32; ++j) {
        x[j] = row[t + j * 256];
        mx = fmaxf(mx, x[j]);
    }
#pragma unroll
    for (int o = 16; o > 0; o >>= 1) mx = fmaxf(mx, __shfl_xor(mx, o, 32));
    __shared__ float red_max[8];
    __shared__ float red_sum[8];
    if ((t & 31) == 0) red_max[t >> 5] = mx;
    __syncthreads();
#pragma unroll
    for (int j = 0; j < 8; ++j) mx = fmaxf(mx, red_max[j]);

    float sum = 0.f;
#pragma unroll
    for (int j = 0; j < 32; ++j) {
        x[j] = __expf(x[j] - mx);
        sum += x[j];
    }
#pragma unroll
    for (int o = 16; o > 0; o >>= 1) sum += __shfl_xor(sum, o, 32);
    if ((t & 31) == 0) red_sum[t >> 5] = sum;
    __syncthreads();
    float total = 0.f;
#pragma unroll
    for (int j = 0; j < 8; ++j) total += red_sum[j];

    float inv = 1.0f / total;
#pragma unroll
    for (int j = 0; j < 32; ++j) row[t + j * 256] = x[j] * inv;
}

extern "C" void kernel_launch(void* const* d_in, const int* in_sizes, int n_in,
                              void* d_out, int out_size, void* d_ws, size_t ws_size,
                              hipStream_t stream) {
    const float* mc = (const float*)d_in[0];   // [B,H]
    const float* sn = (const float*)d_in[1];   // [B,H,L]
    const float* v  = (const float*)d_in[2];   // [1,1,H]
    const float* W  = (const float*)d_in[3];   // [1,H,2H]
    float* out = (float*)d_out;                // [B,1,L]

    uint32_t* w1p  = (uint32_t*)d_ws;                    // 131072 B packed A frags
    float*    bias = (float*)((char*)d_ws + 131072);     // 65536 B bias[b,h]

    pack_w1_kernel<<<16, 256, 0, stream>>>(W, w1p);
    bias_kernel<<<64, 256, 0, stream>>>(W, mc, bias);

    dim3 grid(L_ / NTILE, B_);
    attn_scores_kernel<<<grid, 128, 0, stream>>>(sn, v, bias, (const v16h*)w1p, out);

    softmax_kernel<<<B_, 256, 0, stream>>>(out);
}
